// LSTM_75428215652813
// MI455X (gfx1250) — matrix operand order repaired and checked by reading, not compiled, not run on hardware
//
#include <hip/hip_runtime.h>
#include <stdint.h>

#define B_   512
#define S_   128
#define I_   160
#define H_   256
#define G_   1024   /* 4H */
#define KC_  416    /* I + H, = 13*32 */
#define KI_  13     /* K iterations for LSTM step */
#define LIN_ 2368   /* = 74*32 */
#define MLP_ 512
#define C_   53
#define ENT_ 68

typedef __attribute__((ext_vector_type(16))) __bf16 v16bf;
typedef __attribute__((ext_vector_type(8)))  float  v8f;
typedef __attribute__((ext_vector_type(4)))  unsigned int u32x4;
typedef __attribute__((ext_vector_type(8)))  int i32x8;
typedef __attribute__((ext_vector_type(4)))  int i32x4;

__device__ __forceinline__ unsigned short f2bf(float f) {
  unsigned u = __float_as_uint(f);
  unsigned r = (u + 0x7fffu + ((u >> 16) & 1u)) >> 16;  // RNE
  return (unsigned short)r;
}
__device__ __forceinline__ float bf2f(unsigned short u) {
  return __uint_as_float(((unsigned)u) << 16);
}
// branch-free activations (v_exp_f32 + v_rcp_f32; no libm EXEC-mask branches)
__device__ __forceinline__ float sigf(float x) {
  return __fdividef(1.f, 1.f + __expf(-x));
}
__device__ __forceinline__ float tanh_fast(float x) {
  return 1.f - __fdividef(2.f, __expf(2.f * x) + 1.f);
}

// ---- WMMA fragment loaders (CDNA5 wave32 layouts, 16x16x32 bf16) ----
// A: 16(M) x 32(K), row-major source with row stride lda (elements).
// lanes 0-15: row=lane, K = {0..7, 16..23}; lanes 16-31: row=lane-16, K = {8..15, 24..31}
// Works for both global and LDS-resident sources (addrspace inferred).
__device__ __forceinline__ v16bf load_a_frag(const unsigned short* A, int lda) {
  int lane = threadIdx.x & 31;
  int hf   = lane >> 4;
  int row  = lane & 15;
  const unsigned short* p = A + (size_t)row * lda + 8 * hf;
  v16bf a;
#pragma unroll
  for (int v = 0; v < 8; ++v) {
    int k = (v < 4) ? (2 * v) : (8 + 2 * v);  // v>=4 -> 16 + 2*(v-4)
    union { unsigned u; __bf16 h[2]; } cv;
    cv.u = *(const unsigned*)(p + k);
    a[2 * v]     = cv.h[0];
    a[2 * v + 1] = cv.h[1];
  }
  return a;
}
// B: 32(K) x 16(N) where the source W is [N,K] row-major (computes A * W^T).
__device__ __forceinline__ v16bf load_b_frag(const unsigned short* W, int ldw) {
  int lane = threadIdx.x & 31;
  int hf   = lane >> 4;
  int col  = lane & 15;
  const unsigned short* p = W + (size_t)col * ldw + 8 * hf;
  v16bf b;
#pragma unroll
  for (int v = 0; v < 8; ++v) {
    union { unsigned u; __bf16 h[2]; } cv;
    cv.u = *(const unsigned*)(p + ((v < 4) ? (2 * v) : (8 + 2 * v)));
    b[2 * v]     = cv.h[0];
    b[2 * v + 1] = cv.h[1];
  }
  return b;
}

// ---- Tensor Data Mover: 2D tile (rows x dim0 elems of 2B) global -> LDS ----
// D# per CDNA5 ISA; groups 2/3 zero (2D tensor).
// This toolchain (clang-23 / amdgpu-toolchain) uses the 6-arg builtin:
//   (uint32x4 g0, int32x8 g1, int32x4, int32x4, int32x8, i32 cpol)
#if __has_builtin(__builtin_amdgcn_tensor_load_to_lds)
#define HAVE_TDM 1
__device__ __forceinline__ void tdm_load_2d(unsigned lds_off, const void* gaddr,
                                            unsigned dim0, unsigned rows,
                                            unsigned stride_elems) {
  unsigned long long ga = (unsigned long long)gaddr;
  u32x4 g0;
  g0[0] = 1u;                                   // count=1, user descriptor
  g0[1] = lds_off;                              // lds_addr
  g0[2] = (unsigned)(ga & 0xFFFFFFFFu);         // global_addr[31:0]
  g0[3] = (unsigned)((ga >> 32) & 0x01FFFFFFu)  // global_addr[56:32]
          | (2u << 30);                         // type=2 (image)
  i32x8 g1;
  g1[0] = 0x00010000;                           // wg_mask=0, data_size=1 (2B)
  g1[1] = (int)((dim0 & 0xFFFFu) << 16);        // tensor_dim0[15:0] @ bits 63:48
  g1[2] = (int)((dim0 >> 16) | ((rows & 0xFFFFu) << 16)); // dim0 hi | tensor_dim1 lo
  g1[3] = (int)(((rows >> 16) & 0xFFFFu) | ((dim0 & 0xFFFFu) << 16)); // dim1 hi | tile_dim0
  g1[4] = (int)(rows & 0xFFFFu);                // tile_dim1=rows, tile_dim2=0
  g1[5] = (int)stride_elems;                    // tensor_dim0_stride[31:0]
  g1[6] = 0;                                    // stride hi | dim1_stride lo
  g1[7] = 0;
  i32x4 z4 = {0, 0, 0, 0};
  i32x8 z8 = {0, 0, 0, 0, 0, 0, 0, 0};
  __builtin_amdgcn_tensor_load_to_lds(g0, g1, z4, z4, z8, 0);
}
#else
#define HAVE_TDM 0
#endif

__device__ __forceinline__ unsigned lds_offset_of(const void* p) {
  return (unsigned)(unsigned long long)p;  // LDS aperture: low 32 bits = offset
}

// ---------------- prep kernels ----------------
__global__ void pack_wcat_kernel(const float* __restrict__ Wih,
                                 const float* __restrict__ Whh,
                                 unsigned short* __restrict__ Wcat) {
  int gid = blockIdx.x * blockDim.x + threadIdx.x;
  if (gid >= G_ * KC_) return;
  int k = gid % KC_, n = gid / KC_;
  float v = (k < I_) ? Wih[(size_t)n * I_ + k] : Whh[(size_t)n * H_ + (k - I_)];
  Wcat[gid] = f2bf(v);
}

__global__ void bias_pack_kernel(const float* __restrict__ bih,
                                 const float* __restrict__ bhh,
                                 float* __restrict__ bc) {
  int gid = blockIdx.x * blockDim.x + threadIdx.x;
  if (gid < G_) bc[gid] = bih[gid] + bhh[gid];
}

__global__ void cvt_bf16_kernel(const float* __restrict__ src,
                                unsigned short* __restrict__ dst, int n) {
  int gid = blockIdx.x * blockDim.x + threadIdx.x;
  if (gid < n) dst[gid] = f2bf(src[gid]);
}

__global__ void embed_kernel(const int* __restrict__ word, const int* __restrict__ tag,
                             const int* __restrict__ pos1, const int* __restrict__ pos2,
                             const float* __restrict__ we, const float* __restrict__ te,
                             const float* __restrict__ p1e, const float* __restrict__ p2e,
                             unsigned short* __restrict__ xbf) {
  int gid = blockIdx.x * blockDim.x + threadIdx.x;
  if (gid >= B_ * S_ * I_) return;
  int c = gid % I_;
  int bs = gid / I_;
  float v;
  if (c < 100)       v = we[(size_t)word[bs] * 100 + c];
  else if (c < 120)  v = te[(size_t)tag[bs] * 20 + (c - 100)];
  else if (c < 140)  v = p1e[(size_t)pos1[bs] * 20 + (c - 120)];
  else               v = p2e[(size_t)pos2[bs] * 20 + (c - 140)];
  xbf[gid] = f2bf(v);
}

__global__ void find_idx_kernel(const int* __restrict__ pos1, const int* __restrict__ pos2,
                                int* __restrict__ idx1, int* __restrict__ idx2) {
  int b = blockIdx.x * blockDim.x + threadIdx.x;
  if (b >= B_) return;
  int i1 = 0, i2 = 0;
  for (int s = S_ - 1; s >= 0; --s) {
    if (pos1[b * S_ + s] == ENT_) i1 = s;
    if (pos2[b * S_ + s] == ENT_) i2 = s;
  }
  idx1[b] = i1;
  idx2[b] = i2;
}

// ---------------- fused LSTM step (both directions) ----------------
// All 8 waves of a block share one 16-row batch tile: its A data (x_t || h_prev,
// 13.3KB) is staged into LDS once via the Tensor Data Mover, then each wave
// computes one 16x16 hidden tile with 4 accumulators at gate offsets
// {0,256,512,768} so the cell update is fully in-register. B (weights) stream
// from L2 with double-buffered fragments overlapping the WMMAs.
__global__ void lstm_step_kernel(const unsigned short* __restrict__ xbf,
                                 const unsigned short* __restrict__ Wf,
                                 const unsigned short* __restrict__ Wb,
                                 const float* __restrict__ biasf,
                                 const float* __restrict__ biasb,
                                 const unsigned short* __restrict__ hprev_f,
                                 unsigned short* __restrict__ hnext_f,
                                 const unsigned short* __restrict__ hprev_b,
                                 unsigned short* __restrict__ hnext_b,
                                 float* __restrict__ c_f, float* __restrict__ c_b,
                                 float* __restrict__ outbuf, int t) {
  const int dir = blockIdx.y;
  const unsigned short* W     = dir ? Wb : Wf;
  const float* bias           = dir ? biasb : biasf;
  const unsigned short* hprev = dir ? hprev_b : hprev_f;
  unsigned short* hnext       = dir ? hnext_b : hnext_f;
  float* cbuf                 = dir ? c_b : c_f;
  const int t_eff = dir ? (S_ - 1 - t) : t;

  int wave = threadIdx.x >> 5;
  int widx = blockIdx.x * 8 + wave;  // 0..511
  int m0 = (widx >> 4) * 16;         // batch-row tile base (same for all waves in block)
  int n0 = (widx & 15) * 16;         // hidden-col tile base (0..240)

  const unsigned short* xrow = xbf + ((size_t)m0 * S_ + t_eff) * I_;
  const unsigned short* hrow = hprev + (size_t)m0 * H_;

  __shared__ unsigned short ax[16 * I_];  // 16 rows x 160 (x_t slice)
  __shared__ unsigned short ah[16 * H_];  // 16 rows x 256 (h_prev slice)

#if HAVE_TDM
  if (wave == 0) {
    tdm_load_2d(lds_offset_of(ax), xrow, I_, 16, S_ * I_);
    tdm_load_2d(lds_offset_of(ah), hrow, H_, 16, H_);
    __builtin_amdgcn_s_wait_tensorcnt(0);
  }
  __syncthreads();
#else
  for (int i = threadIdx.x; i < 16 * (I_ * 2 / 16); i += 256) {  // 16B chunks
    int r = i / (I_ * 2 / 16), cc = i % (I_ * 2 / 16);
    ((uint4*)ax)[r * (I_ * 2 / 16) + cc] =
        ((const uint4*)(xrow + (size_t)r * S_ * I_))[cc];
  }
  for (int i = threadIdx.x; i < 16 * (H_ * 2 / 16); i += 256) {
    int r = i / (H_ * 2 / 16), cc = i % (H_ * 2 / 16);
    ((uint4*)ah)[r * (H_ * 2 / 16) + cc] =
        ((const uint4*)(hrow + (size_t)r * H_))[cc];
  }
  __syncthreads();
#endif

  // prefetch next timestep's activation rows (global_prefetch_b8)
  {
    int t_next = dir ? (t_eff - 1) : (t_eff + 1);
    int lane = threadIdx.x & 31;
    if (t_next >= 0 && t_next < S_ && lane < 16) {
      __builtin_prefetch(xbf + ((size_t)(m0 + lane) * S_ + t_next) * I_, 0, 3);
    }
  }

  v8f acc[4] = {};
  v16bf b_cur[4];
#pragma unroll
  for (int j = 0; j < 4; ++j)
    b_cur[j] = load_b_frag(W + (size_t)(j * H_ + n0) * KC_, KC_);

#pragma unroll
  for (int kk = 0; kk < KI_; ++kk) {
    int kc = kk * 32;
    v16bf a_cur = (kc < I_) ? load_a_frag(ax + kc, I_)
                            : load_a_frag(ah + (kc - I_), H_);
    v16bf b_nxt[4];
    if (kk + 1 < KI_) {
      int kn = kc + 32;
#pragma unroll
      for (int j = 0; j < 4; ++j)
        b_nxt[j] = load_b_frag(W + (size_t)(j * H_ + n0) * KC_ + kn, KC_);
    }
#pragma unroll
    for (int j = 0; j < 4; ++j)
      acc[j] = __builtin_amdgcn_wmma_f32_16x16x32_bf16(
          false, a_cur, false, b_cur[j], (short)0, acc[j], false, false);
    if (kk + 1 < KI_) {
#pragma unroll
      for (int j = 0; j < 4; ++j) b_cur[j] = b_nxt[j];
    }
  }

  int lane = threadIdx.x & 31;
  int hf = lane >> 4;
  int col = n0 + (lane & 15);
  float bi = bias[0 * H_ + col], bf = bias[1 * H_ + col];
  float bg = bias[2 * H_ + col], bo = bias[3 * H_ + col];
#pragma unroll
  for (int v = 0; v < 8; ++v) {
    int brow = m0 + 8 * hf + v;
    float gi = acc[0][v] + bi;
    float gf = acc[1][v] + bf;
    float gg = acc[2][v] + bg;
    float go = acc[3][v] + bo;
    size_t ci = (size_t)brow * H_ + col;
    float cn = sigf(gf) * cbuf[ci] + sigf(gi) * tanh_fast(gg);
    float hn = sigf(go) * tanh_fast(cn);
    cbuf[ci] = cn;
    hnext[ci] = f2bf(hn);
    outbuf[((size_t)brow * S_ + t_eff) * (2 * H_) + dir * H_ + col] = hn;
  }
}

// ---------------- feature assembly ----------------
// feat layout: [m1 512 | m2 512 | e1 160 | fe1 512 | e2 160 | fe2 512] = 2368
__global__ void feature_kernel(const float* __restrict__ outbuf,
                               const unsigned short* __restrict__ xbf,
                               const int* __restrict__ idx1, const int* __restrict__ idx2,
                               unsigned short* __restrict__ feat) {
  int b = blockIdx.x;
  int i1 = idx1[b], i2 = idx2[b];
  unsigned short* fb = feat + (size_t)b * LIN_;
  for (int c = threadIdx.x; c < 2 * H_; c += blockDim.x) {
    float m1 = -3.4e38f, m2 = -3.4e38f;
    for (int t = 0; t < S_; ++t) {
      float v = outbuf[((size_t)b * S_ + t) * (2 * H_) + c];
      if (t < i2) m1 = fmaxf(m1, v);
      if (t >= i1) m2 = fmaxf(m2, v);
    }
    if (i2 == 0) m1 = outbuf[((size_t)b * S_) * (2 * H_) + c];
    fb[c]        = f2bf(m1);
    fb[512 + c]  = f2bf(m2);
    fb[1184 + c] = f2bf(outbuf[((size_t)b * S_ + i1) * (2 * H_) + c]);  // fe1
    fb[1856 + c] = f2bf(outbuf[((size_t)b * S_ + i2) * (2 * H_) + c]);  // fe2
  }
  for (int c = threadIdx.x; c < I_; c += blockDim.x) {
    fb[1024 + c] = xbf[((size_t)b * S_ + i1) * I_ + c];  // e1
    fb[1696 + c] = xbf[((size_t)b * S_ + i2) * I_ + c];  // e2
  }
}

// ---------------- MLP layer 1 (WMMA, relu+bias, pipelined K loop) ----------------
__global__ void mlp1_kernel(const unsigned short* __restrict__ feat,
                            const unsigned short* __restrict__ W1bf,
                            const float* __restrict__ b1,
                            unsigned short* __restrict__ h1) {
  int wave = threadIdx.x >> 5;
  int widx = blockIdx.x * 8 + wave;  // 0..255
  int m0 = (widx >> 3) * 16;         // 0..496
  int n0 = (widx & 7) * 64;          // 0..448

  const unsigned short* arow = feat + (size_t)m0 * LIN_;

  v8f acc[4] = {};
  v16bf a_cur = load_a_frag(arow, LIN_);
  v16bf b_cur[4];
#pragma unroll
  for (int j = 0; j < 4; ++j)
    b_cur[j] = load_b_frag(W1bf + (size_t)(n0 + j * 16) * LIN_, LIN_);

#pragma unroll 2
  for (int kc = 0; kc < LIN_; kc += 32) {
    int kn = kc + 32;
    v16bf a_nxt;
    v16bf b_nxt[4];
    bool has = kn < LIN_;
    if (has) {
      a_nxt = load_a_frag(arow + kn, LIN_);
#pragma unroll
      for (int j = 0; j < 4; ++j)
        b_nxt[j] = load_b_frag(W1bf + (size_t)(n0 + j * 16) * LIN_ + kn, LIN_);
    }
#pragma unroll
    for (int j = 0; j < 4; ++j)
      acc[j] = __builtin_amdgcn_wmma_f32_16x16x32_bf16(
          false, a_cur, false, b_cur[j], (short)0, acc[j], false, false);
    if (has) {
      a_cur = a_nxt;
#pragma unroll
      for (int j = 0; j < 4; ++j) b_cur[j] = b_nxt[j];
    }
  }
  int lane = threadIdx.x & 31;
  int hf = lane >> 4;
  int colL = lane & 15;
#pragma unroll
  for (int j = 0; j < 4; ++j) {
    int col = n0 + j * 16 + colL;
    float bb = b1[col];
#pragma unroll
    for (int v = 0; v < 8; ++v) {
      int row = m0 + 8 * hf + v;
      float x = acc[j][v] + bb;
      h1[(size_t)row * MLP_ + col] = f2bf(x > 0.f ? x : 0.f);
    }
  }
}

// ---------------- MLP layer 2 (tiny, VALU) ----------------
__global__ void mlp2_kernel(const unsigned short* __restrict__ h1,
                            const float* __restrict__ W2, const float* __restrict__ b2,
                            float* __restrict__ out) {
  int b = blockIdx.x;
  int c = threadIdx.x;
  if (c >= C_) return;
  float acc = b2[c];
  const unsigned short* hp = h1 + (size_t)b * MLP_;
  const float* wp = W2 + (size_t)c * MLP_;
  for (int k = 0; k < MLP_; ++k) acc += bf2f(hp[k]) * wp[k];
  out[b * C_ + c] = acc;
}

// ---------------- launch ----------------
extern "C" void kernel_launch(void* const* d_in, const int* in_sizes, int n_in,
                              void* d_out, int out_size, void* d_ws, size_t ws_size,
                              hipStream_t stream) {
  (void)in_sizes; (void)n_in; (void)out_size; (void)ws_size;
  const int* word = (const int*)d_in[0];
  const int* tag  = (const int*)d_in[1];
  const int* pos1 = (const int*)d_in[2];
  const int* pos2 = (const int*)d_in[3];
  const float* word_emb = (const float*)d_in[4];
  const float* tag_emb  = (const float*)d_in[5];
  const float* pos1_emb = (const float*)d_in[6];
  const float* pos2_emb = (const float*)d_in[7];
  const float* Wih_f = (const float*)d_in[8];
  const float* Whh_f = (const float*)d_in[9];
  const float* bih_f = (const float*)d_in[10];
  const float* bhh_f = (const float*)d_in[11];
  const float* Wih_b = (const float*)d_in[12];
  const float* Whh_b = (const float*)d_in[13];
  const float* bih_b = (const float*)d_in[14];
  const float* bhh_b = (const float*)d_in[15];
  const float* W1 = (const float*)d_in[16];
  const float* b1 = (const float*)d_in[17];
  const float* W2 = (const float*)d_in[18];
  const float* b2 = (const float*)d_in[19];

  char* p = (char*)d_ws;
  auto alloc = [&](size_t bytes) -> void* {
    void* r = (void*)p;
    p += (bytes + 255) & ~(size_t)255;
    return r;
  };
  unsigned short* xbf   = (unsigned short*)alloc((size_t)B_ * S_ * I_ * 2);
  unsigned short* WcatF = (unsigned short*)alloc((size_t)G_ * KC_ * 2);
  unsigned short* WcatB = (unsigned short*)alloc((size_t)G_ * KC_ * 2);
  float* biasF = (float*)alloc(G_ * 4);
  float* biasB = (float*)alloc(G_ * 4);
  unsigned short* h_f[2], *h_b[2];
  h_f[0] = (unsigned short*)alloc((size_t)B_ * H_ * 2);
  h_f[1] = (unsigned short*)alloc((size_t)B_ * H_ * 2);
  h_b[0] = (unsigned short*)alloc((size_t)B_ * H_ * 2);
  h_b[1] = (unsigned short*)alloc((size_t)B_ * H_ * 2);
  float* c_f  = (float*)alloc((size_t)B_ * H_ * 4);
  float* c_b  = (float*)alloc((size_t)B_ * H_ * 4);
  float* outb = (float*)alloc((size_t)B_ * S_ * 2 * H_ * 4);
  int* idx1 = (int*)alloc(B_ * 4);
  int* idx2 = (int*)alloc(B_ * 4);
  unsigned short* feat = (unsigned short*)alloc((size_t)B_ * LIN_ * 2);
  unsigned short* W1bf = (unsigned short*)alloc((size_t)MLP_ * LIN_ * 2);
  unsigned short* h1   = (unsigned short*)alloc((size_t)B_ * MLP_ * 2);

  (void)hipMemsetAsync(c_f, 0, (size_t)B_ * H_ * 4, stream);
  (void)hipMemsetAsync(c_b, 0, (size_t)B_ * H_ * 4, stream);
  (void)hipMemsetAsync(h_f[0], 0, (size_t)B_ * H_ * 2, stream);
  (void)hipMemsetAsync(h_b[0], 0, (size_t)B_ * H_ * 2, stream);

  int nW = G_ * KC_;
  pack_wcat_kernel<<<(nW + 255) / 256, 256, 0, stream>>>(Wih_f, Whh_f, WcatF);
  pack_wcat_kernel<<<(nW + 255) / 256, 256, 0, stream>>>(Wih_b, Whh_b, WcatB);
  bias_pack_kernel<<<(G_ + 255) / 256, 256, 0, stream>>>(bih_f, bhh_f, biasF);
  bias_pack_kernel<<<(G_ + 255) / 256, 256, 0, stream>>>(bih_b, bhh_b, biasB);
  int n1 = MLP_ * LIN_;
  cvt_bf16_kernel<<<(n1 + 255) / 256, 256, 0, stream>>>(W1, W1bf, n1);

  int nE = B_ * S_ * I_;
  embed_kernel<<<(nE + 255) / 256, 256, 0, stream>>>(word, tag, pos1, pos2, word_emb,
                                                     tag_emb, pos1_emb, pos2_emb, xbf);
  find_idx_kernel<<<(B_ + 255) / 256, 256, 0, stream>>>(pos1, pos2, idx1, idx2);

  for (int t = 0; t < S_; ++t) {
    int pr = t & 1, nx = pr ^ 1;
    lstm_step_kernel<<<dim3(64, 2), 256, 0, stream>>>(
        xbf, WcatF, WcatB, biasF, biasB,
        h_f[pr], h_f[nx], h_b[pr], h_b[nx], c_f, c_b, outb, t);
  }

  feature_kernel<<<B_, 256, 0, stream>>>(outb, xbf, idx1, idx2, feat);
  mlp1_kernel<<<32, 256, 0, stream>>>(feat, W1bf, b1, h1);
  mlp2_kernel<<<B_, 64, 0, stream>>>(h1, W2, b2, (float*)d_out);
}
